// ScaledDotProductAttention_78743930405429
// MI455X (gfx1250) — matrix operand order repaired and checked by reading, not compiled, not run on hardware
//
#include <hip/hip_runtime.h>

typedef __attribute__((ext_vector_type(16))) _Float16 v16h;
typedef __attribute__((ext_vector_type(8)))  float    v8f;

#define Bn   4
#define Hn   16
#define Sn   2048
#define Dn   64
#define QT   64          // query rows per workgroup
#define KT   32          // key columns per iteration
#define NWAV 4           // waves per workgroup (128 threads, wave32)
#define NITER (Sn / KT)  // 64

#define KPAD 72          // halves per K/Q LDS row (64 + 8 pad), 144B, 16B aligned
#define VPAD 40          // halves per Vt/P LDS row  (32 + 8 pad), 80B, 16B aligned
#define NEG_INF_F (-1e10f)

union Frag  { uint4 u[2]; v16h v; };
union Pack4 { _Float16 h[4]; uint2 u; };
union H16   { _Float16 h; unsigned short s; };

// CDNA5 async global->LDS copy (ASYNCcnt-tracked), inline asm for toolchain
// portability per the bridge doc. ldsOff = byte offset within workgroup LDS
// (low 32 bits of the generic pointer to a __shared__ object), ga = 64-bit
// global address. One B128 (16 bytes) per lane.
__device__ __forceinline__ void async_ld_b128(unsigned ldsOff, unsigned long long ga) {
    asm volatile("global_load_async_to_lds_b128 %0, %1, off"
                 :: "v"(ldsOff), "v"(ga) : "memory");
}
__device__ __forceinline__ void wait_async_le1() {
    asm volatile("s_wait_asynccnt 0x1" ::: "memory");
}
__device__ __forceinline__ void wait_async_le0() {
    asm volatile("s_wait_asynccnt 0x0" ::: "memory");
}

__global__ __launch_bounds__(128)
void flash_attn_f16_wmma(const float* __restrict__ Q,
                         const float* __restrict__ K,
                         const float* __restrict__ V,
                         const unsigned char* __restrict__ M,
                         float* __restrict__ out)
{
    __shared__ __align__(16) unsigned short Qlds[QT * KPAD];        // 9216 B, f16 (pre-scaled by 1/8)
    __shared__ __align__(16) unsigned short Klds[KT * KPAD];        // 4608 B, f16 row-major [key][d]
    __shared__ __align__(16) unsigned short Vtld[Dn * VPAD];        // 5120 B, f16 transposed [d][key]
    __shared__ __align__(16) unsigned short Plds[NWAV * 16 * VPAD]; // 5120 B, per-wave prob scratch
    __shared__ __align__(16) unsigned char  Mlds[2][QT * KT];       // 4096 B, double-buffered mask bytes

    const int t    = threadIdx.x;
    const int w    = t >> 5;          // wave id 0..3
    const int lane = t & 31;
    const int half = lane >> 4;       // 0: lanes 0-15, 1: lanes 16-31
    const int ln   = lane & 15;

    const int head = blockIdx.y;                  // 0..63 (b*H + h)
    const int q0   = blockIdx.x * QT;             // query block start

    const float* Qh = Q + (size_t)head * Sn * Dn;
    const float* Kh = K + (size_t)head * Sn * Dn;
    const float* Vh = V + (size_t)head * Sn * Dn;
    const unsigned char* Mh = M + (size_t)head * Sn * Sn;

    // Per-thread slice of each 64x32-byte mask tile: row = t>>1, 16B half-row.
    const int mrow_ld  = t >> 1;
    const int mpart_ld = (t & 1) * 16;
    const unsigned long long mga_base =
        (unsigned long long)(Mh + (size_t)(q0 + mrow_ld) * Sn + mpart_ld);
    const unsigned mlds_base =
        (unsigned)(size_t)&Mlds[0][mrow_ld * KT + mpart_ld];

    // Kick off async mask prefetch of tile 0 (overlaps with Q staging).
    async_ld_b128(mlds_base, mga_base);

    // ---- Stage Q tile (64x64) into LDS as f16, scaled by 1/8 -----------------
    #pragma unroll
    for (int i = 0; i < 8; ++i) {
        int flat = t + 128 * i;
        int row  = flat >> 4;
        int c4   = flat & 15;
        float4 qv = reinterpret_cast<const float4*>(Qh)[(size_t)(q0 + row) * (Dn / 4) + c4];
        Pack4 p;
        p.h[0] = (_Float16)(qv.x * 0.125f);
        p.h[1] = (_Float16)(qv.y * 0.125f);
        p.h[2] = (_Float16)(qv.z * 0.125f);
        p.h[3] = (_Float16)(qv.w * 0.125f);
        *reinterpret_cast<uint2*>(&Qlds[row * KPAD + c4 * 4]) = p.u;
    }
    __syncthreads();

    // ---- Build this wave's Q A-fragments (16 rows x 64 d -> two 16x32 frags) --
    const int lr = w * 16 + ln;       // workgroup-local query row for A operand
    Frag qa[2];
    #pragma unroll
    for (int f = 0; f < 2; ++f) {
        int base = lr * KPAD + f * 32 + half * 8;
        qa[f].u[0] = *reinterpret_cast<const uint4*>(&Qlds[base]);
        qa[f].u[1] = *reinterpret_cast<const uint4*>(&Qlds[base + 16]);
    }

    // ---- Online-softmax running state (row = r + 8*half, col = ln) -----------
    float mrun[8], lrun[8];
    #pragma unroll
    for (int r = 0; r < 8; ++r) { mrun[r] = -INFINITY; lrun[r] = 0.0f; }
    v8f o0 = {}, o1 = {}, o2 = {}, o3 = {};

    unsigned short* Pw = &Plds[w * 16 * VPAD];

    for (int kb = 0; kb < NITER; ++kb) {
        const int k0 = kb * KT;

        // ---- Async mask prefetch for next tile into the other buffer --------
        if (kb + 1 < NITER) {
            async_ld_b128(mlds_base + ((kb + 1) & 1) * (QT * KT),
                          mga_base + (unsigned long long)(k0 + KT));
        }

        // ---- Cooperative K/V tile loads (f32 -> f16 conversion) -------------
        #pragma unroll
        for (int i = 0; i < 4; ++i) {
            int flat = t + 128 * i;
            int row  = flat >> 4;
            int c4   = flat & 15;
            float4 kv = reinterpret_cast<const float4*>(Kh)[(size_t)(k0 + row) * (Dn / 4) + c4];
            Pack4 p;
            p.h[0] = (_Float16)kv.x; p.h[1] = (_Float16)kv.y;
            p.h[2] = (_Float16)kv.z; p.h[3] = (_Float16)kv.w;
            *reinterpret_cast<uint2*>(&Klds[row * KPAD + c4 * 4]) = p.u;
        }
        #pragma unroll
        for (int i = 0; i < 4; ++i) {
            int flat = t + 128 * i;
            int row  = flat >> 4;     // key index in tile
            int c4   = flat & 15;
            float4 vv = reinterpret_cast<const float4*>(Vh)[(size_t)(k0 + row) * (Dn / 4) + c4];
            H16 hh;
            hh.h = (_Float16)vv.x; Vtld[(c4 * 4 + 0) * VPAD + row] = hh.s;
            hh.h = (_Float16)vv.y; Vtld[(c4 * 4 + 1) * VPAD + row] = hh.s;
            hh.h = (_Float16)vv.z; Vtld[(c4 * 4 + 2) * VPAD + row] = hh.s;
            hh.h = (_Float16)vv.w; Vtld[(c4 * 4 + 3) * VPAD + row] = hh.s;
        }
        // L2 prefetch of next K/V tiles (global_prefetch_b8)
        if (kb + 1 < NITER) {
            const char* nk = (const char*)(Kh + (size_t)(k0 + KT) * Dn);
            const char* nv = (const char*)(Vh + (size_t)(k0 + KT) * Dn);
            __builtin_prefetch(nk + t * 64, 0, 0);
            __builtin_prefetch(nv + t * 64, 0, 0);
        }

        // Current tile's async mask copy must have landed in LDS (in-order
        // completion; the newest prefetch may remain outstanding).
        if (kb + 1 < NITER) wait_async_le1(); else wait_async_le0();
        __syncthreads();

        // ---- Scores: S(16x32) = Qtile(16x64) @ Ktile^T(64x32), pre-scaled ----
        v8f s0 = {}, s1 = {};
        #pragma unroll
        for (int j = 0; j < 2; ++j) {          // key n-chunk (cols 16j..16j+15)
            #pragma unroll
            for (int f = 0; f < 2; ++f) {      // d k-chunk (32f..32f+31)
                Frag kbf;
                int base = (16 * j + ln) * KPAD + f * 32 + half * 8;
                kbf.u[0] = *reinterpret_cast<const uint4*>(&Klds[base]);
                kbf.u[1] = *reinterpret_cast<const uint4*>(&Klds[base + 16]);
                if (j == 0)
                    s0 = __builtin_amdgcn_wmma_f32_16x16x32_f16(false, qa[f].v, false, kbf.v,
                                                                (short)0, s0, false, false);
                else
                    s1 = __builtin_amdgcn_wmma_f32_16x16x32_f16(false, qa[f].v, false, kbf.v,
                                                                (short)0, s1, false, false);
            }
        }

        // ---- Mask + online softmax (per C-fragment row r+8*half) ------------
        const unsigned char* Mbuf = Mlds[kb & 1];
        #pragma unroll
        for (int r = 0; r < 8; ++r) {
            int mrow = w * 16 + r + 8 * half;
            float a0 = s0[r], a1 = s1[r];
            if (Mbuf[mrow * KT + ln])      a0 = NEG_INF_F;
            if (Mbuf[mrow * KT + 16 + ln]) a1 = NEG_INF_F;
            float tmax = fmaxf(a0, a1);
            tmax = fmaxf(tmax, __shfl_xor(tmax, 1, 32));
            tmax = fmaxf(tmax, __shfl_xor(tmax, 2, 32));
            tmax = fmaxf(tmax, __shfl_xor(tmax, 4, 32));
            tmax = fmaxf(tmax, __shfl_xor(tmax, 8, 32));
            float nm = fmaxf(mrun[r], tmax);
            float al = __expf(mrun[r] - nm);
            mrun[r] = nm;
            float p0 = __expf(a0 - nm);
            float p1 = __expf(a1 - nm);
            float rs = p0 + p1;
            rs += __shfl_xor(rs, 1, 32);
            rs += __shfl_xor(rs, 2, 32);
            rs += __shfl_xor(rs, 4, 32);
            rs += __shfl_xor(rs, 8, 32);
            lrun[r] = lrun[r] * al + rs;
            H16 hp;
            hp.h = (_Float16)p0; Pw[(r + 8 * half) * VPAD + ln]      = hp.s;
            hp.h = (_Float16)p1; Pw[(r + 8 * half) * VPAD + 16 + ln] = hp.s;
            o0[r] *= al; o1[r] *= al; o2[r] *= al; o3[r] *= al;
        }

        // ---- P A-fragment (16x32) from per-wave LDS scratch ------------------
        Frag pa;
        {
            int base = ln * VPAD + half * 8;
            pa.u[0] = *reinterpret_cast<const uint4*>(&Pw[base]);
            pa.u[1] = *reinterpret_cast<const uint4*>(&Pw[base + 16]);
        }

        // ---- O(16x64) += P(16x32) @ Vtile(32x64) ----------------------------
        #pragma unroll
        for (int c = 0; c < 4; ++c) {
            Frag vb;
            int base = (16 * c + ln) * VPAD + half * 8;
            vb.u[0] = *reinterpret_cast<const uint4*>(&Vtld[base]);
            vb.u[1] = *reinterpret_cast<const uint4*>(&Vtld[base + 16]);
            if (c == 0) o0 = __builtin_amdgcn_wmma_f32_16x16x32_f16(false, pa.v, false, vb.v, (short)0, o0, false, false);
            if (c == 1) o1 = __builtin_amdgcn_wmma_f32_16x16x32_f16(false, pa.v, false, vb.v, (short)0, o1, false, false);
            if (c == 2) o2 = __builtin_amdgcn_wmma_f32_16x16x32_f16(false, pa.v, false, vb.v, (short)0, o2, false, false);
            if (c == 3) o3 = __builtin_amdgcn_wmma_f32_16x16x32_f16(false, pa.v, false, vb.v, (short)0, o3, false, false);
        }
        __syncthreads();   // protect LDS tiles before next iteration's stores
    }

    // ---- Epilogue: out = O / l ----------------------------------------------
    #pragma unroll
    for (int r = 0; r < 8; ++r) {
        float inv = 1.0f / lrun[r];
        size_t row = (size_t)head * Sn * Dn + (size_t)(q0 + w * 16 + r + 8 * half) * Dn;
        out[row + 0 * 16 + ln] = o0[r] * inv;
        out[row + 1 * 16 + ln] = o1[r] * inv;
        out[row + 2 * 16 + ln] = o2[r] * inv;
        out[row + 3 * 16 + ln] = o3[r] * inv;
    }
}

extern "C" void kernel_launch(void* const* d_in, const int* in_sizes, int n_in,
                              void* d_out, int out_size, void* d_ws, size_t ws_size,
                              hipStream_t stream) {
    (void)in_sizes; (void)n_in; (void)d_ws; (void)ws_size; (void)out_size;
    const float* Q = (const float*)d_in[0];
    const float* K = (const float*)d_in[1];
    const float* V = (const float*)d_in[2];
    const unsigned char* M = (const unsigned char*)d_in[3];  // jnp.bool_ -> 1 byte/elem
    float* out = (float*)d_out;

    dim3 grid(Sn / QT, Bn * Hn);   // (32, 64)
    dim3 block(128);               // 4 waves (wave32)
    flash_attn_f16_wmma<<<grid, block, 0, stream>>>(Q, K, V, M, out);
}
